// biEncodeLSTM_52939766891157
// MI455X (gfx1250) — matrix operand order repaired and checked by reading, not compiled, not run on hardware
//
#include <hip/hip_runtime.h>

// ---------------------------------------------------------------------------
// Bidirectional 2-layer LSTM for MI455X (gfx1250, wave32)
// WMMA f16 16x16x32 for gate GEMMs + TDM (tensor_load_to_lds) pipelined
// staging of the layer-1 input slice. B=32, T=1024, H=256.
// ---------------------------------------------------------------------------

typedef _Float16 v16h __attribute__((ext_vector_type(16)));
typedef _Float16 v8h  __attribute__((ext_vector_type(8)));
typedef float    v8f  __attribute__((ext_vector_type(8)));
typedef unsigned int u32x4 __attribute__((ext_vector_type(4)));
typedef int          i32x4 __attribute__((ext_vector_type(4)));
typedef int          i32x8 __attribute__((ext_vector_type(8)));

#define Bsz 32
#define Tsz 1024
#define Hsz 256
#define Gsz 1024          // 4H
#define K0  256           // layer0 recurrent K
#define K1  768           // layer1 fused K (512 input + 256 recurrent)
#define RS0 272           // LDS row stride (halves), layer0 h buffer
#define RSX 528           // LDS row stride (halves), layer1 x buffer (512+16)
#define RSH 272           // LDS row stride (halves), layer1 h buffer

#if __has_builtin(__builtin_amdgcn_tensor_load_to_lds) && \
    __has_builtin(__builtin_amdgcn_s_wait_tensorcnt)
#define USE_TDM 1
#else
#define USE_TDM 0
#endif

static __device__ inline float frcp(float x) {
#if __has_builtin(__builtin_amdgcn_rcpf)
  return __builtin_amdgcn_rcpf(x);   // single v_rcp_f32
#else
  return 1.0f / x;
#endif
}
static __device__ inline float sigm(float x) { return frcp(1.0f + __expf(-x)); }
static __device__ inline float ftanh(float x) {
  // tanh(x) = 1 - 2/(e^(2x)+1)
  return 1.0f - 2.0f * frcp(1.0f + __expf(2.0f * x));
}
static __device__ inline v8f splat8(float s) {
  v8f v = {s, s, s, s, s, s, s, s};
  return v;
}

static __device__ inline v8f wmma16(v16h a, v16h b, v8f c) {
  return __builtin_amdgcn_wmma_f32_16x16x32_f16(
      false, a, false, b, (short)0, c, false, false);
}

union FragU { v16h v; v8h h[2]; };

// A fragment (16x32 f16) from LDS rows of `rs` halves.
static __device__ inline v16h ldA(const _Float16* buf, int rs, int mtile,
                                  int kbase, int lane) {
  const int row = mtile * 16 + (lane & 15);
  const int k0  = kbase + ((lane & 16) ? 16 : 0);
  FragU u;
  u.h[0] = *(const v8h*)(buf + row * rs + k0);
  u.h[1] = *(const v8h*)(buf + row * rs + k0 + 8);
  return u.v;
}

// B fragment (32x16 f16) from weights packed (N,K) row-major f16:
// 16 contiguous halves per lane = two 16B loads.
static __device__ inline v16h ldB(const _Float16* __restrict__ W, int kstride,
                                  int nbase, int kbase, int lane) {
  const int col = nbase + (lane & 15);
  const int k0  = kbase + ((lane & 16) ? 16 : 0);
  const v8h* p = (const v8h*)(W + (size_t)col * kstride + k0);
  FragU u;
  u.h[0] = p[0];
  u.h[1] = p[1];
  return u.v;
}

#if USE_TDM
// Issue TENSOR_LOAD_TO_LDS: 32 rows x 1024B (f16) tile from h1, row stride
// T*512 elements, LDS pad 32B per row (-> row stride 528 halves).
static __device__ inline void tdm_load_x(const _Float16* gsrc, unsigned lds_byte) {
  unsigned long long ga = (unsigned long long)gsrc;
  u32x4 g0;
  g0[0] = 1u;                                    // count=1, user descriptor
  g0[1] = lds_byte;                              // lds_addr (bytes)
  g0[2] = (unsigned)(ga & 0xFFFFFFFFull);        // global_addr[31:0]
  g0[3] = (unsigned)((ga >> 32) & 0x01FFFFFFull) // global_addr[56:32]
          | (2u << 30);                          // type=2 (image)
  i32x8 g1;
  g1[0] = (1 << 16)        // data_size = 1 -> 2 bytes
        | (1 << 20)        // pad_enable
        | (7 << 22)        // pad_interval: 256 DWORDs = 1024B (one row)
        | (7 << 25);       // pad_amount: 8 DWORDs = 32B
  g1[1] = (512 << 16);     // tensor_dim0[15:0]=512 (atomic_barrier_addr=0)
  g1[2] = (32 << 16);      // tensor_dim0 hi=0 | tensor_dim1[15:0]=32
  g1[3] = (512 << 16);     // tensor_dim1 hi=0 | tile_dim0=512
  g1[4] = 32;              // tile_dim1=32, tile_dim2=0
  g1[5] = (int)(Tsz * 512);// tensor_dim0_stride[31:0] = 524288 elems
  g1[6] = 0;               // stride hi / dim1_stride (unused, 2D tile)
  g1[7] = 0;
  i32x4 z4 = {0, 0, 0, 0};
#if __has_include(<hip/amd_detail/amd_gfx1250_TDM.h>)
  i32x8 z8 = {0, 0, 0, 0, 0, 0, 0, 0};
  __builtin_amdgcn_tensor_load_to_lds(g0, g1, z4, z4, z8, 0);
#else
  __builtin_amdgcn_tensor_load_to_lds(g0, g1, z4, z4, 0);
#endif
}
#endif

// ---------------------------------------------------------------------------
// Weight packing: dst[n*(Ki+Kh) + k] = f16( k<Ki ? Wi[k][n] : Wh[k-Ki][n] )
// ---------------------------------------------------------------------------
__global__ void pack_w_kernel(const float* __restrict__ Wi, int Ki,
                              const float* __restrict__ Wh, int Kh,
                              _Float16* __restrict__ dst) {
  const int K = Ki + Kh;
  const int idx = blockIdx.x * blockDim.x + threadIdx.x;
  if (idx >= Gsz * K) return;
  const int n = idx / K;
  const int k = idx % K;
  float v = (k < Ki) ? Wi[(size_t)k * Gsz + n] : Wh[(size_t)(k - Ki) * Gsz + n];
  dst[idx] = (_Float16)v;
}

// ---------------------------------------------------------------------------
// Layer 0 (input dim 1): gates = x_t*Wi0 + h@Wh0 + b (x folded into epilogue)
// ---------------------------------------------------------------------------
__global__ __launch_bounds__(512) void lstm_l0_kernel(
    const float* __restrict__ x,
    const _Float16* __restrict__ Wt0f, const _Float16* __restrict__ Wt0r,
    const float* __restrict__ Wi0f, const float* __restrict__ Wi0r,
    const float* __restrict__ b0f, const float* __restrict__ b0r,
    _Float16* __restrict__ h1) {
  const int dir = blockIdx.x;
  const _Float16* Wt = dir ? Wt0r : Wt0f;
  const float* Wi0   = dir ? Wi0r : Wi0f;
  const float* bb    = dir ? b0r  : b0f;

  __shared__ __align__(16) _Float16 aBuf[Bsz * RS0];
  __shared__ float xsh[Bsz];

  const int tid   = threadIdx.x;
  const int lane  = tid & 31;
  const int wid   = tid >> 5;
  const int n0    = wid * 16;
  const int nl    = lane & 15;
  const int mhalf = (lane & 16) ? 8 : 0;

  float biasv[4], wiv[4];
#pragma unroll
  for (int q = 0; q < 4; ++q) {
    biasv[q] = bb[q * Hsz + n0 + nl];
    wiv[q]   = Wi0[q * Hsz + n0 + nl];
  }

  float cacc[2][8] = {};
  float hval[2][8] = {};

  for (int t = 0; t < Tsz; ++t) {
    const int tt = dir ? (Tsz - 1 - t) : t;

    __syncthreads();
#pragma unroll
    for (int mt = 0; mt < 2; ++mt)
#pragma unroll
      for (int r = 0; r < 8; ++r) {
        const int m = mt * 16 + r + mhalf;
        aBuf[m * RS0 + n0 + nl] = (_Float16)hval[mt][r];
      }
    if (tid < Bsz) xsh[tid] = x[(size_t)tid * Tsz + tt];
    __syncthreads();

#pragma unroll
    for (int mt = 0; mt < 2; ++mt) {
      v8f acc[4];
#pragma unroll
      for (int q = 0; q < 4; ++q) acc[q] = splat8(biasv[q]);

#pragma unroll 1
      for (int kt = 0; kt < K0 / 32; ++kt) {
        v16h a = ldA(aBuf, RS0, mt, kt * 32, lane);
#pragma unroll
        for (int q = 0; q < 4; ++q) {
          v16h bf = ldB(Wt, K0, q * Hsz + n0, kt * 32, lane);
          acc[q] = wmma16(a, bf, acc[q]);
        }
      }

#pragma unroll
      for (int r = 0; r < 8; ++r) {
        const int m = mt * 16 + r + mhalf;
        const float xv = xsh[m];
        const float gi = acc[0][r] + xv * wiv[0];
        const float gf = acc[1][r] + xv * wiv[1];
        const float gg = acc[2][r] + xv * wiv[2];
        const float go = acc[3][r] + xv * wiv[3];
        float c = sigm(gf) * cacc[mt][r] + sigm(gi) * ftanh(gg);
        cacc[mt][r] = c;
        float h = sigm(go) * ftanh(c);
        hval[mt][r] = h;
        h1[(size_t)(m * Tsz + tt) * 512 + dir * Hsz + n0 + nl] = (_Float16)h;
      }
    }
  }
}

// ---------------------------------------------------------------------------
// Layer 1 (fused): gates = [x_t | h](32x768) @ W(768x1024) + b
// x_t = h1[:, tt, :] staged by TDM one step ahead (double-buffered LDS).
// ---------------------------------------------------------------------------
__global__ __launch_bounds__(512) void lstm_l1_kernel(
    const _Float16* __restrict__ h1,
    const _Float16* __restrict__ Wt1f, const _Float16* __restrict__ Wt1r,
    const float* __restrict__ b1f, const float* __restrict__ b1r,
    float* __restrict__ out) {
  const int dir = blockIdx.x;
  const _Float16* Wt = dir ? Wt1r : Wt1f;
  const float* bb    = dir ? b1r  : b1f;

  __shared__ __align__(16) _Float16 xBuf[2][Bsz * RSX];  // x_t slices (double buf)
  __shared__ __align__(16) _Float16 hBuf[Bsz * RSH];     // recurrent h

  const int tid   = threadIdx.x;
  const int lane  = tid & 31;
  const int wid   = tid >> 5;
  const int n0    = wid * 16;
  const int nl    = lane & 15;
  const int mhalf = (lane & 16) ? 8 : 0;

  float biasv[4];
#pragma unroll
  for (int q = 0; q < 4; ++q) biasv[q] = bb[q * Hsz + n0 + nl];

  float cacc[2][8] = {};
  float hval[2][8] = {};

  float* c_out = out;                 // (B, 512)
  float* h_out = out + Bsz * 512;     // (B, 512)
  float* y     = out + 2 * Bsz * 512; // (B, T, 512)

#if USE_TDM
  const unsigned ldsx0 = (unsigned)(unsigned long long)(const void*)&xBuf[0][0];
  const unsigned ldsx1 = (unsigned)(unsigned long long)(const void*)&xBuf[1][0];
  if (wid == 0) {  // prefetch t=0 slice
    const int tt0 = dir ? (Tsz - 1) : 0;
    tdm_load_x(h1 + (size_t)tt0 * 512, ldsx0);
  }
#else
  const int crow = tid >> 4;          // 0..31
  const int ccol = (tid & 15) * 32;   // 0..480
#endif

  for (int t = 0; t < Tsz; ++t) {
    const int tt = dir ? (Tsz - 1 - t) : t;
#if USE_TDM
    const int cur = t & 1;
    if (wid == 0) __builtin_amdgcn_s_wait_tensorcnt(0);  // xBuf[cur] landed
#else
    const int cur = 0;
#endif
    __syncthreads();  // publish TDM data / prior-step LDS reads complete

#pragma unroll
    for (int mt = 0; mt < 2; ++mt)
#pragma unroll
      for (int r = 0; r < 8; ++r) {
        const int m = mt * 16 + r + mhalf;
        hBuf[m * RSH + n0 + nl] = (_Float16)hval[mt][r];
      }

#if USE_TDM
    if (wid == 0 && t + 1 < Tsz) {  // stage next step's slice into other buffer
      const int tn = dir ? (Tsz - 2 - t) : (t + 1);
      tdm_load_x(h1 + (size_t)tn * 512, (t & 1) ? ldsx0 : ldsx1);
    }
#else
    {  // synchronous staging fallback
      const v8h* src = (const v8h*)(h1 + (size_t)(crow * Tsz + tt) * 512 + ccol);
      v8h* dst = (v8h*)(&xBuf[0][0] + crow * RSX + ccol);
      dst[0] = src[0]; dst[1] = src[1]; dst[2] = src[2]; dst[3] = src[3];
    }
#endif
    __syncthreads();

#pragma unroll
    for (int mt = 0; mt < 2; ++mt) {
      v8f acc[4];
#pragma unroll
      for (int q = 0; q < 4; ++q) acc[q] = splat8(biasv[q]);

      // K 0..511 : x part from xBuf[cur]
#pragma unroll 1
      for (int kt = 0; kt < 16; ++kt) {
        v16h a = ldA(&xBuf[cur][0], RSX, mt, kt * 32, lane);
#pragma unroll
        for (int q = 0; q < 4; ++q) {
          v16h bf = ldB(Wt, K1, q * Hsz + n0, kt * 32, lane);
          acc[q] = wmma16(a, bf, acc[q]);
        }
      }
      // K 512..767 : recurrent part from hBuf
#pragma unroll 1
      for (int kt = 0; kt < 8; ++kt) {
        v16h a = ldA(hBuf, RSH, mt, kt * 32, lane);
#pragma unroll
        for (int q = 0; q < 4; ++q) {
          v16h bf = ldB(Wt, K1, q * Hsz + n0, 512 + kt * 32, lane);
          acc[q] = wmma16(a, bf, acc[q]);
        }
      }

#pragma unroll
      for (int r = 0; r < 8; ++r) {
        const int m = mt * 16 + r + mhalf;
        float c = sigm(acc[1][r]) * cacc[mt][r] + sigm(acc[0][r]) * ftanh(acc[2][r]);
        cacc[mt][r] = c;
        float h = sigm(acc[3][r]) * ftanh(c);
        hval[mt][r] = h;
        y[(size_t)(m * Tsz + tt) * 512 + dir * Hsz + n0 + nl] = h;
      }
    }
  }

  // final carries (layer-1 states, concat fwd|rev)
#pragma unroll
  for (int mt = 0; mt < 2; ++mt)
#pragma unroll
    for (int r = 0; r < 8; ++r) {
      const int m = mt * 16 + r + mhalf;
      c_out[m * 512 + dir * Hsz + n0 + nl] = cacc[mt][r];
      h_out[m * 512 + dir * Hsz + n0 + nl] = hval[mt][r];
    }
}

// ---------------------------------------------------------------------------
extern "C" void kernel_launch(void* const* d_in, const int* in_sizes, int n_in,
                              void* d_out, int out_size, void* d_ws, size_t ws_size,
                              hipStream_t stream) {
  const float* x    = (const float*)d_in[0];
  const float* Wi0f = (const float*)d_in[1];
  const float* Wh0f = (const float*)d_in[2];
  const float* b0f  = (const float*)d_in[3];
  const float* Wi0r = (const float*)d_in[4];
  const float* Wh0r = (const float*)d_in[5];
  const float* b0r  = (const float*)d_in[6];
  const float* Wi1f = (const float*)d_in[7];
  const float* Wh1f = (const float*)d_in[8];
  const float* b1f  = (const float*)d_in[9];
  const float* Wi1r = (const float*)d_in[10];
  const float* Wh1r = (const float*)d_in[11];
  const float* b1r  = (const float*)d_in[12];

  char* ws = (char*)d_ws;
  _Float16* h1   = (_Float16*)ws;                                  // 32 MB
  _Float16* Wt0f = (_Float16*)(ws + (size_t)Bsz * Tsz * 512 * 2);  // 512 KB each
  _Float16* Wt0r = Wt0f + (size_t)Gsz * K0;
  _Float16* Wt1f = Wt0r + (size_t)Gsz * K0;                        // 1.5 MB each
  _Float16* Wt1r = Wt1f + (size_t)Gsz * K1;

  {
    int n0 = Gsz * K0, n1 = Gsz * K1;
    pack_w_kernel<<<(n0 + 255) / 256, 256, 0, stream>>>(Wi0f, 0, Wh0f, K0, Wt0f);
    pack_w_kernel<<<(n0 + 255) / 256, 256, 0, stream>>>(Wi0r, 0, Wh0r, K0, Wt0r);
    pack_w_kernel<<<(n1 + 255) / 256, 256, 0, stream>>>(Wi1f, 512, Wh1f, Hsz, Wt1f);
    pack_w_kernel<<<(n1 + 255) / 256, 256, 0, stream>>>(Wi1r, 512, Wh1r, Hsz, Wt1r);
  }

  lstm_l0_kernel<<<2, 512, 0, stream>>>(x, Wt0f, Wt0r, Wi0f, Wi0r, b0f, b0r, h1);
  lstm_l1_kernel<<<2, 512, 0, stream>>>(h1, Wt1f, Wt1r, b1f, b1r, (float*)d_out);
}
